// MultiHeadedAttention_31112743092510
// MI455X (gfx1250) — matrix operand order repaired and checked by reading, not compiled, not run on hardware
//
#include <hip/hip_runtime.h>
#include <hip/hip_bf16.h>

typedef __attribute__((ext_vector_type(16))) _Float16 v16h;
typedef __attribute__((ext_vector_type(8)))  _Float16 v8h;
typedef __attribute__((ext_vector_type(4)))  _Float16 v4h;
typedef __attribute__((ext_vector_type(8)))  float    v8f;

#define B_   2
#define S_   2048
#define D_   1024
#define H_   16
#define HD_  64
#define M_   (B_ * S_)       // 4096
#define BM   128
#define BN   64
#define BK   32
#define LDSP 40              // padded LDS row stride in halves (conflict-free b128)
#define ABUF_H 5120          // A tile: 128 rows * 40 halves
#define BBUF_H 2560          // B tile:  64 rows * 40 halves
#define BUF_H  (ABUF_H + BBUF_H)   // halves per double-buffer slot (15360 B)

// ---------------------------------------------------------------------------
// WMMA helpers
// ---------------------------------------------------------------------------
static __device__ __forceinline__ v8f wmma_f16(v16h a, v16h b, v8f c) {
  return __builtin_amdgcn_wmma_f32_16x16x32_f16(
      false, a, false, b, (short)0, c, false, false);
}

// A-operand fragment (16 rows x 32 K): lane = row + 16*kg; halves 0..7 ->
// k = 8*kg..8*kg+7, halves 8..15 -> k = 8*kg+16..8*kg+23 (ISA 7.12.2).
static __device__ __forceinline__ v16h ld_a_frag(const _Float16* p, int stride, int lane) {
  const int row = lane & 15;
  const int kg  = lane >> 4;
  const _Float16* base = p + (size_t)row * stride + kg * 8;
  v8h lo = *(const v8h*)(base);
  v8h hi = *(const v8h*)(base + 16);
  v16h f;
#pragma unroll
  for (int i = 0; i < 8; ++i) { f[i] = lo[i]; f[i + 8] = hi[i]; }
  return f;
}

// B-operand fragment (32 K x 16 cols): lane = col + 16*kg; halves 0..7 ->
// k = 8*kg..8*kg+7, halves 8..15 -> k = 8*kg+16..8*kg+23; source rows are B columns.
static __device__ __forceinline__ v16h ld_b_frag(const _Float16* p, int stride, int lane) {
  const int col = lane & 15;
  const int kg  = lane >> 4;
  const _Float16* base = p + (size_t)col * stride + kg * 8;
  v8h lo = *(const v8h*)(base);
  v8h hi = *(const v8h*)(base + 16);
  v16h f;
#pragma unroll
  for (int i = 0; i < 8; ++i) { f[i] = lo[i]; f[i + 8] = hi[i]; }
  return f;
}

// ---------------------------------------------------------------------------
// Async global -> LDS copy (16 B per lane), tracked by ASYNCcnt.
// ---------------------------------------------------------------------------
static __device__ __forceinline__ void async_cp16(unsigned int lds_addr,
                                                  unsigned int voff,
                                                  unsigned long long sbase) {
  asm volatile("global_load_async_to_lds_b128 %0, %1, %2"
               :: "v"(lds_addr), "v"(voff), "s"(sbase) : "memory");
}
static __device__ __forceinline__ void wait_async0() {
  asm volatile("s_wait_asynccnt 0x0" ::: "memory");
}

// ---------------------------------------------------------------------------
// fp32 -> f16 conversion (elementwise, float4 wide)
// ---------------------------------------------------------------------------
__global__ __launch_bounds__(256)
void cvt_f16_kernel(const float* __restrict__ src, _Float16* __restrict__ dst, int n4) {
  const int i = blockIdx.x * 256 + threadIdx.x;
  if (i < n4) {
    const float4 f = ((const float4*)src)[i];
    v4h h; h[0] = (_Float16)f.x; h[1] = (_Float16)f.y;
           h[2] = (_Float16)f.z; h[3] = (_Float16)f.w;
    ((v4h*)dst)[i] = h;
  }
}

// ---------------------------------------------------------------------------
// Unified f16 GEMM: C = A[M_,D_] @ Bw[D_,D_]^T + bias
//   omode 0: f16 out, [B,H,S,HD]   (Q, K)
//   omode 1: f16 out, [B,H,HD,S]   (V^T)
//   omode 2: f32 out, [M_,D_]      (final projection)
// 128 threads (4 waves); BM=128 x BN=64; double-buffered async LDS staging.
// ---------------------------------------------------------------------------
__global__ __launch_bounds__(128)
void gemm_f16_kernel(const _Float16* __restrict__ A, const _Float16* __restrict__ Bw,
                     const float* __restrict__ bias, void* __restrict__ outp,
                     int omode) {
  __shared__ _Float16 smem[2 * BUF_H];           // 30720 B
  const int tid  = threadIdx.x;
  const int lane = tid & 31;
  const int wave = tid >> 5;
  const int m0 = blockIdx.x * BM;
  const int n0 = blockIdx.y * BN;

  // Per-thread staging addresses (constant across K steps except base buffer).
  const unsigned int lds0 = (unsigned int)(uintptr_t)(&smem[0]);
  unsigned int ldsA[4], vofA[4], ldsB[2], vofB[2];
#pragma unroll
  for (int u = 0; u < 4; ++u) {                  // A: 128 rows x 4 segs of 16B
    const int idx = tid + u * 128;
    const int r = idx >> 2, seg = idx & 3;
    ldsA[u] = (unsigned int)(r * (LDSP * 2) + seg * 16);
    vofA[u] = (unsigned int)(r * (D_ * 2) + seg * 16);
  }
#pragma unroll
  for (int u = 0; u < 2; ++u) {                  // B: 64 rows x 4 segs of 16B
    const int idx = tid + u * 128;
    const int r = idx >> 2, seg = idx & 3;
    ldsB[u] = (unsigned int)(ABUF_H * 2 + r * (LDSP * 2) + seg * 16);
    vofB[u] = (unsigned int)(r * (D_ * 2) + seg * 16);
  }

  const _Float16* Abase = A  + (size_t)m0 * D_;
  const _Float16* Bbase = Bw + (size_t)n0 * D_;

  // Prologue: stage tile kk=0 into buffer 0.
  {
    const unsigned long long sa = (unsigned long long)(uintptr_t)Abase;
    const unsigned long long sb = (unsigned long long)(uintptr_t)Bbase;
#pragma unroll
    for (int u = 0; u < 4; ++u) async_cp16(lds0 + ldsA[u], vofA[u], sa);
#pragma unroll
    for (int u = 0; u < 2; ++u) async_cp16(lds0 + ldsB[u], vofB[u], sb);
  }

  v8f acc[2][4] = {};
  int buf = 0;

  for (int kk = 0; kk < D_; kk += BK) {
    wait_async0();            // this wave's copies for current buffer done
    __syncthreads();          // all waves' copies landed; prev reads finished

    if (kk + BK < D_) {       // overlap: stage next tile into the other buffer
      const unsigned int bo = (unsigned int)((buf ^ 1) * (BUF_H * 2));
      const unsigned long long sa =
          (unsigned long long)(uintptr_t)(Abase + kk + BK);
      const unsigned long long sb =
          (unsigned long long)(uintptr_t)(Bbase + kk + BK);
#pragma unroll
      for (int u = 0; u < 4; ++u) async_cp16(lds0 + bo + ldsA[u], vofA[u], sa);
#pragma unroll
      for (int u = 0; u < 2; ++u) async_cp16(lds0 + bo + ldsB[u], vofB[u], sb);
    }

    const _Float16* Abuf = smem + buf * BUF_H;
    const _Float16* Bbuf = Abuf + ABUF_H;
    const v16h a0 = ld_a_frag(Abuf + (wave * 32) * LDSP,      LDSP, lane);
    const v16h a1 = ld_a_frag(Abuf + (wave * 32 + 16) * LDSP, LDSP, lane);
    v16h b[4];
#pragma unroll
    for (int t = 0; t < 4; ++t) b[t] = ld_b_frag(Bbuf + (t * 16) * LDSP, LDSP, lane);
#pragma unroll
    for (int t = 0; t < 4; ++t) {
      acc[0][t] = wmma_f16(a0, b[t], acc[0][t]);
      acc[1][t] = wmma_f16(a1, b[t], acc[1][t]);
    }
    __syncthreads();          // reads of this buffer complete before re-stage
    buf ^= 1;
  }

  // Epilogue: C layout -> lane n = lane&15, rows (lane>>4)*8 + r.
  const int nl = lane & 15;
#pragma unroll
  for (int s = 0; s < 2; ++s) {
    const int mb = wave * 32 + s * 16 + (lane >> 4) * 8;
#pragma unroll
    for (int t = 0; t < 4; ++t) {
      const int nIdx = n0 + t * 16 + nl;
      const float bvv = bias[nIdx];
      const int hh = nIdx >> 6;      // / HD
      const int dd = nIdx & 63;      // % HD
#pragma unroll
      for (int r = 0; r < 8; ++r) {
        const int mIdx = m0 + mb + r;
        const float val = acc[s][t][r] + bvv;
        if (omode == 2) {
          ((float*)outp)[(size_t)mIdx * D_ + nIdx] = val;
        } else {
          const int bb = mIdx >> 11;   // / S
          const int ss = mIdx & 2047;  // % S
          _Float16* o16 = (_Float16*)outp;
          if (omode == 0)
            o16[(((size_t)(bb * H_ + hh)) * S_ + ss) * HD_ + dd] = (_Float16)val;
          else
            o16[(((size_t)(bb * H_ + hh)) * HD_ + dd) * S_ + ss] = (_Float16)val;
        }
      }
    }
  }
}

// ---------------------------------------------------------------------------
// Flash attention: one wave per 16-row Q tile per (b,h); 32-row K/V chunks,
// online softmax, mask applied per reference.
// ---------------------------------------------------------------------------
__global__ __launch_bounds__(32)
void attn_kernel(const _Float16* __restrict__ Q16,  // [B,H,S,HD]
                 const _Float16* __restrict__ K16,  // [B,H,S,HD]
                 const _Float16* __restrict__ Vt16, // [B,H,HD,S]
                 const int* __restrict__ mask,      // [B,1,S,S]
                 _Float16* __restrict__ AO) {       // [B,S,H*HD]
  __shared__ _Float16 Pl[16 * LDSP];
  const int lane  = threadIdx.x;
  const int qbase = blockIdx.x * 16;
  const int h     = blockIdx.y;
  const int b     = blockIdx.z;
  const size_t bh = (size_t)b * H_ + h;

  const _Float16* Qp = Q16 + (bh * S_ + qbase) * HD_;
  const _Float16* Kp = K16 + bh * S_ * HD_;
  const _Float16* Vp = Vt16 + bh * HD_ * S_;
  const int* mrowbase = mask + ((size_t)b * S_ + qbase) * S_;

  const int nl = lane & 15;
  const int mb = (lane >> 4) * 8;

  const v16h qa0 = ld_a_frag(Qp,      HD_, lane);
  const v16h qa1 = ld_a_frag(Qp + 32, HD_, lane);

  float mrun[8], lrun[8];
  v8f O[4] = {};
#pragma unroll
  for (int r = 0; r < 8; ++r) { mrun[r] = -3.0e38f; lrun[r] = 0.f; }

  const float inv_scale = 0.125f;    // 1/sqrt(HD)

  for (int kk = 0; kk < S_; kk += 32) {
    v8f c0 = {}, c1 = {};
    {
      const _Float16* k0 = Kp + (size_t)kk * HD_;
      const _Float16* k1 = Kp + (size_t)(kk + 16) * HD_;
      const v16h kb00 = ld_b_frag(k0,      HD_, lane);
      const v16h kb01 = ld_b_frag(k0 + 32, HD_, lane);
      const v16h kb10 = ld_b_frag(k1,      HD_, lane);
      const v16h kb11 = ld_b_frag(k1 + 32, HD_, lane);
      c0 = wmma_f16(qa0, kb00, c0);
      c0 = wmma_f16(qa1, kb01, c0);
      c1 = wmma_f16(qa0, kb10, c1);
      c1 = wmma_f16(qa1, kb11, c1);
    }
    float s0[8], s1[8];
#pragma unroll
    for (int r = 0; r < 8; ++r) {
      const int* mrow = mrowbase + (size_t)(mb + r) * S_ + kk;
      s0[r] = (mrow[nl]      != 0) ? c0[r] * inv_scale : -1e9f;
      s1[r] = (mrow[nl + 16] != 0) ? c1[r] * inv_scale : -1e9f;
    }
    float mnew[8], corr[8];
#pragma unroll
    for (int r = 0; r < 8; ++r) {
      float t = fmaxf(s0[r], s1[r]);
      t = fmaxf(t, __shfl_xor(t, 1, 32));
      t = fmaxf(t, __shfl_xor(t, 2, 32));
      t = fmaxf(t, __shfl_xor(t, 4, 32));
      t = fmaxf(t, __shfl_xor(t, 8, 32));
      mnew[r] = fmaxf(mrun[r], t);
      corr[r] = __expf(mrun[r] - mnew[r]);
      mrun[r] = mnew[r];
    }
    float p0[8], p1[8];
#pragma unroll
    for (int r = 0; r < 8; ++r) {
      p0[r] = __expf(s0[r] - mnew[r]);
      p1[r] = __expf(s1[r] - mnew[r]);
      float rs = p0[r] + p1[r];
      rs += __shfl_xor(rs, 1, 32);
      rs += __shfl_xor(rs, 2, 32);
      rs += __shfl_xor(rs, 4, 32);
      rs += __shfl_xor(rs, 8, 32);
      lrun[r] = lrun[r] * corr[r] + rs;
      Pl[(mb + r) * LDSP + nl]      = (_Float16)p0[r];
      Pl[(mb + r) * LDSP + nl + 16] = (_Float16)p1[r];
    }
#pragma unroll
    for (int t = 0; t < 4; ++t)
#pragma unroll
      for (int r = 0; r < 8; ++r) O[t][r] *= corr[r];

    const v16h pa = ld_a_frag(Pl, LDSP, lane);
#pragma unroll
    for (int t = 0; t < 4; ++t) {
      const v16h vb = ld_b_frag(Vp + (size_t)(t * 16) * S_ + kk, S_, lane);
      O[t] = wmma_f16(pa, vb, O[t]);
    }
  }

  _Float16* outp = AO + ((size_t)b * S_ + qbase) * D_ + h * HD_;
#pragma unroll
  for (int t = 0; t < 4; ++t)
#pragma unroll
    for (int r = 0; r < 8; ++r)
      outp[(size_t)(mb + r) * D_ + t * 16 + nl] = (_Float16)(O[t][r] / lrun[r]);
}

// ---------------------------------------------------------------------------
extern "C" void kernel_launch(void* const* d_in, const int* in_sizes, int n_in,
                              void* d_out, int out_size, void* d_ws, size_t ws_size,
                              hipStream_t stream) {
  const float* q    = (const float*)d_in[0];
  const float* k    = (const float*)d_in[1];
  const float* v    = (const float*)d_in[2];
  const int*   mask = (const int*)  d_in[3];
  const float* Wq   = (const float*)d_in[4];
  const float* bq   = (const float*)d_in[5];
  const float* Wk   = (const float*)d_in[6];
  const float* bk   = (const float*)d_in[7];
  const float* Wv   = (const float*)d_in[8];
  const float* bv   = (const float*)d_in[9];
  const float* Wo   = (const float*)d_in[10];
  const float* bo   = (const float*)d_in[11];
  float* out = (float*)d_out;

  _Float16* ws = (_Float16*)d_ws;
  const size_t NX = (size_t)M_ * D_;              // 4,194,304 halves (8 MB)
  const size_t NW = (size_t)D_ * D_;              // 1,048,576 halves (2 MB)
  _Float16* Q16  = ws;                            // [B,H,S,HD]
  _Float16* K16  = ws + NX;                       // [B,H,S,HD]
  _Float16* Vt16 = ws + 2 * NX;                   // [B,H,HD,S]
  _Float16* X16q = ws + 3 * NX;                   // also reused as AO16
  _Float16* AO16 = X16q;                          // alias (X16q dead by then)
  _Float16* X16k = ws + 4 * NX;
  _Float16* X16v = ws + 5 * NX;
  _Float16* W16q = ws + 6 * NX;
  _Float16* W16k = W16q + NW;
  _Float16* W16v = W16k + NW;
  _Float16* W16o = W16v + NW;

  // fp32 -> f16 conversions (bandwidth-trivial)
  {
    const int nX4 = (int)(NX / 4), nW4 = (int)(NW / 4);
    const int gX = (nX4 + 255) / 256, gW = (nW4 + 255) / 256;
    cvt_f16_kernel<<<gX, 256, 0, stream>>>(q,  X16q, nX4);
    cvt_f16_kernel<<<gX, 256, 0, stream>>>(k,  X16k, nX4);
    cvt_f16_kernel<<<gX, 256, 0, stream>>>(v,  X16v, nX4);
    cvt_f16_kernel<<<gW, 256, 0, stream>>>(Wq, W16q, nW4);
    cvt_f16_kernel<<<gW, 256, 0, stream>>>(Wk, W16k, nW4);
    cvt_f16_kernel<<<gW, 256, 0, stream>>>(Wv, W16v, nW4);
    cvt_f16_kernel<<<gW, 256, 0, stream>>>(Wo, W16o, nW4);
  }

  dim3 ggrid(M_ / BM, D_ / BN);                    // 32 x 16
  gemm_f16_kernel<<<ggrid, 128, 0, stream>>>(X16q, W16q, bq, (void*)Q16,  0);
  gemm_f16_kernel<<<ggrid, 128, 0, stream>>>(X16k, W16k, bk, (void*)K16,  0);
  gemm_f16_kernel<<<ggrid, 128, 0, stream>>>(X16v, W16v, bv, (void*)Vt16, 1);

  dim3 agrid(S_ / 16, H_, B_);                     // 128 x 16 x 2
  attn_kernel<<<agrid, 32, 0, stream>>>(Q16, K16, Vt16, mask, AO16);

  gemm_f16_kernel<<<ggrid, 128, 0, stream>>>(AO16, W16o, bo, (void*)out, 2);
}
